// Attention_13099650253331
// MI455X (gfx1250) — matrix operand order repaired and checked by reading, not compiled, not run on hardware
//
#include <hip/hip_runtime.h>

// ---------------------------------------------------------------------------
// Flash attention (GQA, causal + sliding window 1024) for MI455X / gfx1250.
// Wave32 WMMA f32_16x16x32_f16, fp32 accumulation, LDS-staged K/V tiles.
// v2: packed LDS staging (b128/b64 stores), mask-free fast path for interior
//     key blocks.
// ---------------------------------------------------------------------------

namespace {
constexpr int kNH  = 32;     // query heads
constexpr int kHS  = 128;    // head size
constexpr int kNKV = 8;      // kv heads
constexpr int kWIN = 1024;   // sliding window
constexpr int kS   = 2048;   // seq len
constexpr float kScale = 0.08838834764831845f;

constexpr int kQStride  = kNH  * kHS;  // 4096 floats per token
constexpr int kKVStride = kNKV * kHS;  // 1024 floats per token

// LDS strides (in halfs), padded to dodge bank conflicts.
constexpr int KSTR = 136;  // K tile row stride: [key][d], 128 + 8 pad
constexpr int VSTR = 40;   // V tile row stride: [d][key], 32 + 8 pad
constexpr int PSTR = 40;   // P scratch row stride: [row][key32], 32 + 8 pad
constexpr float kNegInf = -1e30f;
}  // namespace

typedef __attribute__((ext_vector_type(16))) _Float16 v16h;
typedef __attribute__((ext_vector_type(8)))  _Float16 v8h;
typedef __attribute__((ext_vector_type(4)))  _Float16 v4h;
typedef __attribute__((ext_vector_type(8)))  float    v8f;

__device__ __forceinline__ v16h cat8(v8h lo, v8h hi) {
  return __builtin_shufflevector(lo, hi, 0,1,2,3,4,5,6,7,8,9,10,11,12,13,14,15);
}

__device__ __forceinline__ v8h cvt8(float4 a, float4 b) {
  v8h r;
  r[0]=(_Float16)a.x; r[1]=(_Float16)a.y; r[2]=(_Float16)a.z; r[3]=(_Float16)a.w;
  r[4]=(_Float16)b.x; r[5]=(_Float16)b.y; r[6]=(_Float16)b.z; r[7]=(_Float16)b.w;
  return r;
}

__global__ __launch_bounds__(128, 1)
void gqa_swa_attn_kernel(const float* __restrict__ Q,
                         const float* __restrict__ K,
                         const float* __restrict__ V,
                         float* __restrict__ O) {
  // LDS: K tile 32x128 (row-major), V tile 128x32 (transposed), P scratch.
  __shared__ __align__(16) _Float16 sK[32 * KSTR];        // 8704 B
  __shared__ __align__(16) _Float16 sV[kHS * VSTR];       // 10240 B
  __shared__ __align__(16) _Float16 sP[4 * 16 * PSTR];    // 5120 B

  const int tid  = threadIdx.x;
  const int wv   = tid >> 5;        // wave id 0..3
  const int lane = tid & 31;
  const int g    = lane >> 4;       // half-wave group (0: rows 0-7, 1: rows 8-15)
  const int col  = lane & 15;       // N index within 16-wide tile

  const int qb  = blockIdx.x * 64;  // workgroup query base
  const int h   = blockIdx.y;       // query head
  const int b   = blockIdx.z;       // batch
  const int kvh = h >> 2;           // GQA group of 4
  const int q0  = qb + wv * 16;     // this wave's 16 query rows

  const size_t qBase  = (size_t)b * kS * kQStride;
  const size_t kvBase = (size_t)b * kS * kKVStride;

  // Staging sub-tile for this thread: 4 consecutive keys x 8 consecutive dims.
  const int skq = (tid & 7) * 4;    // key base 0..28
  const int sdq = (tid >> 3) * 8;   // dim base 0..120

  // ---- Load Q as 4 A-fragments (K-chunks of 32 over D=128), pre-scaled ----
  // A 16x32 f16 layout: lanes 0-15 row=lane hold K 0-7 (halves 0-7) and
  // K 16-23 (halves 8-15); lanes 16-31 row=lane-16 hold K 8-15 and K 24-31.
  v16h aq[4];
  {
    const float* qrow = Q + qBase + (size_t)(q0 + col) * kQStride + h * kHS;
    const int koff = g * 8;
#pragma unroll
    for (int kc = 0; kc < 4; ++kc) {
      const float* p = qrow + kc * 32 + koff;
      float4 f0 = *(const float4*)(p);
      float4 f1 = *(const float4*)(p + 4);
      float4 f2 = *(const float4*)(p + 16);
      float4 f3 = *(const float4*)(p + 20);
      v16h a;
      a[0]  = (_Float16)(f0.x * kScale); a[1]  = (_Float16)(f0.y * kScale);
      a[2]  = (_Float16)(f0.z * kScale); a[3]  = (_Float16)(f0.w * kScale);
      a[4]  = (_Float16)(f1.x * kScale); a[5]  = (_Float16)(f1.y * kScale);
      a[6]  = (_Float16)(f1.z * kScale); a[7]  = (_Float16)(f1.w * kScale);
      a[8]  = (_Float16)(f2.x * kScale); a[9]  = (_Float16)(f2.y * kScale);
      a[10] = (_Float16)(f2.z * kScale); a[11] = (_Float16)(f2.w * kScale);
      a[12] = (_Float16)(f3.x * kScale); a[13] = (_Float16)(f3.y * kScale);
      a[14] = (_Float16)(f3.z * kScale); a[15] = (_Float16)(f3.w * kScale);
      aq[kc] = a;
    }
  }

  // Output accumulators: 8 tiles of 16 dims each (C/D 16x16 f32 layout).
  v8f o[8];
#pragma unroll
  for (int n = 0; n < 8; ++n) o[n] = {};

  float mrun[8], lrun[8];
#pragma unroll
  for (int i = 0; i < 8; ++i) { mrun[i] = kNegInf; lrun[i] = 0.0f; }

  // Uniform WG loop over 32-key blocks covering the union of all 4 waves'
  // windows: [qb - 1023, qb + 63], aligned down to 32. Max key = qb+63 <= S-1.
  int jstart = qb - kWIN + 1;
  if (jstart < 0) jstart = 0;
  jstart &= ~31;
  const int jend = qb + 63;

  for (int j0 = jstart; j0 <= jend; j0 += 32) {
    // ---- Cooperative staging (packed stores) ---------------------------
    // K tile [key][d] via b128 row stores; V tile [d][key] via b64 packed
    // 4-key column stores.
    {
      float vbuf[4][8];
#pragma unroll
      for (int kk = 0; kk < 4; ++kk) {
        const int key = skq + kk;
        const float* kr = K + kvBase + (size_t)(j0 + key) * kKVStride + kvh * kHS + sdq;
        const float* vr = V + kvBase + (size_t)(j0 + key) * kKVStride + kvh * kHS + sdq;
        float4 ka = *(const float4*)(kr);
        float4 kb = *(const float4*)(kr + 4);
        *(v8h*)(sK + key * KSTR + sdq) = cvt8(ka, kb);
        float4 va = *(const float4*)(vr);
        float4 vb = *(const float4*)(vr + 4);
        vbuf[kk][0]=va.x; vbuf[kk][1]=va.y; vbuf[kk][2]=va.z; vbuf[kk][3]=va.w;
        vbuf[kk][4]=vb.x; vbuf[kk][5]=vb.y; vbuf[kk][6]=vb.z; vbuf[kk][7]=vb.w;
      }
#pragma unroll
      for (int d = 0; d < 8; ++d) {
        v4h c;
        c[0] = (_Float16)vbuf[0][d];
        c[1] = (_Float16)vbuf[1][d];
        c[2] = (_Float16)vbuf[2][d];
        c[3] = (_Float16)vbuf[3][d];
        *(v4h*)(sV + (sdq + d) * VSTR + skq) = c;
      }
    }
    __syncthreads();

    // Wave-uniform skip: block must overlap this wave's window.
    const bool active = (j0 <= q0 + 15) && (j0 + 31 >= q0 - (kWIN - 1));
    if (active) {
      // ---- Scores: S[16 x 32] = Q(16x128) @ K^T via 8 WMMAs -------------
      v8f c0 = {}, c1 = {};
#pragma unroll
      for (int kc = 0; kc < 4; ++kc) {
        // B 32x16 layout: lane n holds contraction K 0-7,16-23 (lanes 0-15) or
        // 8-15,24-31 (lanes 16-31) of key row `col`.
        const _Float16* kb0 = sK + col * KSTR + kc * 32 + g * 8;
        const _Float16* kb1 = sK + (16 + col) * KSTR + kc * 32 + g * 8;
        v16h b0 = cat8(*(const v8h*)kb0, *(const v8h*)(kb0 + 16));
        v16h b1 = cat8(*(const v8h*)kb1, *(const v8h*)(kb1 + 16));
        c0 = __builtin_amdgcn_wmma_f32_16x16x32_f16(false, aq[kc], false, b0,
                                                    (short)0, c0, false, false);
        c1 = __builtin_amdgcn_wmma_f32_16x16x32_f16(false, aq[kc], false, b1,
                                                    (short)0, c1, false, false);
      }

      // ---- Online softmax (rows live in 16-lane halves) -----------------
      // Interior blocks (valid for all rows/keys of this wave) skip masking.
      float p0[8], p1[8], alpha[8];
      auto softmax_update = [&](bool domask) {
#pragma unroll
        for (int i = 0; i < 8; ++i) {
          float s0, s1;
          bool ok0 = true, ok1 = true;
          if (domask) {
            const int qi  = q0 + g * 8 + i;
            const int ki0 = j0 + col;
            const int ki1 = j0 + 16 + col;
            ok0 = (ki0 <= qi) && (qi - ki0 < kWIN);
            ok1 = (ki1 <= qi) && (qi - ki1 < kWIN);
            s0 = ok0 ? c0[i] : kNegInf;   // scale folded into Q
            s1 = ok1 ? c1[i] : kNegInf;
          } else {
            s0 = c0[i];
            s1 = c1[i];
          }
          float r = fmaxf(s0, s1);
          r = fmaxf(r, __shfl_xor(r, 1, 32));
          r = fmaxf(r, __shfl_xor(r, 2, 32));
          r = fmaxf(r, __shfl_xor(r, 4, 32));
          r = fmaxf(r, __shfl_xor(r, 8, 32));
          const float mn = fmaxf(mrun[i], r);
          alpha[i] = __expf(mrun[i] - mn);
          mrun[i] = mn;
          if (domask) {
            p0[i] = ok0 ? __expf(s0 - mn) : 0.0f;
            p1[i] = ok1 ? __expf(s1 - mn) : 0.0f;
          } else {
            p0[i] = __expf(s0 - mn);
            p1[i] = __expf(s1 - mn);
          }
          float sum = p0[i] + p1[i];
          sum += __shfl_xor(sum, 1, 32);
          sum += __shfl_xor(sum, 2, 32);
          sum += __shfl_xor(sum, 4, 32);
          sum += __shfl_xor(sum, 8, 32);
          lrun[i] = lrun[i] * alpha[i] + sum;
        }
      };
      // Mask-free iff every key in [j0, j0+31] is valid for every row in
      // [q0, q0+15]: causal for row q0 and window for row q0+15.
      const bool fullValid = (j0 + 31 <= q0) && (j0 >= q0 - (kWIN - 16));
      if (fullValid) softmax_update(false);
      else           softmax_update(true);

      // Rescale accumulators by alpha (per row).
#pragma unroll
      for (int n = 0; n < 8; ++n) {
#pragma unroll
        for (int i = 0; i < 8; ++i) o[n][i] *= alpha[i];
      }

      // ---- Re-layout P (C-layout -> A-layout) through per-wave LDS ------
      _Float16* pb = sP + wv * 16 * PSTR;
#pragma unroll
      for (int i = 0; i < 8; ++i) {
        pb[(g * 8 + i) * PSTR + col]      = (_Float16)p0[i];
        pb[(g * 8 + i) * PSTR + 16 + col] = (_Float16)p1[i];
      }
      asm volatile("s_wait_dscnt 0" ::: "memory");  // wave-local LDS RAW
      const _Float16* pr = pb + col * PSTR + g * 8;
      v16h ap = cat8(*(const v8h*)pr, *(const v8h*)(pr + 16));

      // ---- O(16x128) += P(16x32) @ V(32x128): 8 WMMAs -------------------
#pragma unroll
      for (int n = 0; n < 8; ++n) {
        const _Float16* vp = sV + (n * 16 + col) * VSTR + g * 8;
        v16h bv = cat8(*(const v8h*)vp, *(const v8h*)(vp + 16));
        o[n] = __builtin_amdgcn_wmma_f32_16x16x32_f16(false, ap, false, bv,
                                                      (short)0, o[n], false, false);
      }
    }
    __syncthreads();  // protect K/V tiles before next staging round
  }

  // ---- Epilogue: normalize and store ------------------------------------
  float rl[8];
#pragma unroll
  for (int i = 0; i < 8; ++i) rl[i] = (lrun[i] > 0.0f) ? (1.0f / lrun[i]) : 0.0f;
#pragma unroll
  for (int n = 0; n < 8; ++n) {
#pragma unroll
    for (int i = 0; i < 8; ++i) {
      const int row = q0 + g * 8 + i;
      O[qBase + (size_t)row * kQStride + h * kHS + n * 16 + col] = o[n][i] * rl[i];
    }
  }
}

extern "C" void kernel_launch(void* const* d_in, const int* in_sizes, int n_in,
                              void* d_out, int out_size, void* d_ws, size_t ws_size,
                              hipStream_t stream) {
  (void)in_sizes; (void)n_in; (void)out_size; (void)d_ws; (void)ws_size;
  const float* Q = (const float*)d_in[0];
  const float* K = (const float*)d_in[1];
  const float* V = (const float*)d_in[2];
  float* O = (float*)d_out;
  // grid: (S/64 q-blocks, heads, batch); block: 4 waves of 32.
  dim3 grid(kS / 64, kNH, 2);
  dim3 block(128, 1, 1);
  gqa_swa_attn_kernel<<<grid, block, 0, stream>>>(Q, K, V, O);
}
